// PrototypeMemory_44100724195853
// MI455X (gfx1250) — matrix operand order repaired and checked by reading, not compiled, not run on hardware
//
#include <hip/hip_runtime.h>
#include <hip/hip_bf16.h>
#include <math.h>

typedef __attribute__((ext_vector_type(16))) _Float16 v16h;
typedef __attribute__((ext_vector_type(8)))  float    v8f;

#define M_ROWS   262144
#define C_CLS    91
#define C_PAD    96
#define D_DIM    256
#define TAU_INV  10.0f
#define REPC     0.1f

// ---- workspace layout (bytes) ----
#define OFF_SEG      0u         // f32 [91*256]  raw segment sums -> overwritten with new_proto
#define OFF_PROTOSN  93184u     // f32 [91*256]  normalized new prototypes
#define OFF_PROTOH   186368u    // f16 [96*256]  normalized prototypes, f16, zero-padded rows 91..95
#define OFF_RNORM    235520u    // f32 [262144]  1/max(||feat_row||,1e-12)
#define OFF_CNT      1284096u   // i32 [91]      per-class batch counts
#define OFF_WV       1284608u   // f32 [96]      per-class final sample weight (valid*w/mean_w)
#define OFF_VALID    1284992u   // f32 [96]      new_init as float
#define OFF_SCAL     1285376u   // f32 [16]      [0]=pull_acc [1]=rep [2]=1/nv

__device__ __forceinline__ float wred_sum(float v) {
  v += __shfl_xor(v, 16, 32); v += __shfl_xor(v, 8, 32);
  v += __shfl_xor(v, 4, 32);  v += __shfl_xor(v, 2, 32);
  v += __shfl_xor(v, 1, 32);  return v;
}
__device__ __forceinline__ float hred_sum(float v) {   // reduce within 16-lane half
  v += __shfl_xor(v, 8, 32); v += __shfl_xor(v, 4, 32);
  v += __shfl_xor(v, 2, 32); v += __shfl_xor(v, 1, 32); return v;
}
__device__ __forceinline__ float hred_max(float v) {
  v = fmaxf(v, __shfl_xor(v, 8, 32)); v = fmaxf(v, __shfl_xor(v, 4, 32));
  v = fmaxf(v, __shfl_xor(v, 2, 32)); v = fmaxf(v, __shfl_xor(v, 1, 32)); return v;
}

// ---------------- kernel 0: zero accumulators ----------------
__global__ __launch_bounds__(256) void k_zero(char* ws) {
  float* seg  = (float*)(ws + OFF_SEG);
  int*   cnt  = (int*)  (ws + OFF_CNT);
  float* scal = (float*)(ws + OFF_SCAL);
  for (int i = threadIdx.x; i < C_CLS * D_DIM; i += 256) seg[i] = 0.f;
  if (threadIdx.x < C_CLS) cnt[threadIdx.x] = 0;
  if (threadIdx.x < 16)    scal[threadIdx.x] = 0.f;
}

// ---------------- kernel 1: row norms + class segment sums ----------------
// wave-per-row; full (C,D) accumulator staged in 93KB of LDS per workgroup.
__global__ __launch_bounds__(256) void k_accum(const float* __restrict__ feats,
                                               const int* __restrict__ labels,
                                               char* ws) {
  extern __shared__ char smem[];
  float* sseg = (float*)smem;                      // 23296 floats
  int*   scnt = (int*)(smem + C_CLS * D_DIM * 4);  // 96 ints
  float* gseg   = (float*)(ws + OFF_SEG);
  int*   gcnt   = (int*)  (ws + OFF_CNT);
  float* grnorm = (float*)(ws + OFF_RNORM);

  for (int i = threadIdx.x; i < C_CLS * D_DIM; i += 256) sseg[i] = 0.f;
  if (threadIdx.x < C_PAD) scnt[threadIdx.x] = 0;
  __syncthreads();

  const int lane  = threadIdx.x & 31;
  const int gwave = blockIdx.x * 8 + (threadIdx.x >> 5);
  const int nwave = gridDim.x * 8;

  for (int row = gwave; row < M_ROWS; row += nwave) {
    const float* base = feats + (size_t)row * D_DIM + lane * 8;
    const float4 a = *(const float4*)(base);
    const float4 b = *(const float4*)(base + 4);
    float ss = a.x*a.x + a.y*a.y + a.z*a.z + a.w*a.w
             + b.x*b.x + b.y*b.y + b.z*b.z + b.w*b.w;
    ss = wred_sum(ss);
    const float rn = 1.f / fmaxf(sqrtf(ss), 1e-12f);
    if (lane == 0) grnorm[row] = rn;
    const int lab = labels[row];
    float* dst = sseg + lab * D_DIM + lane * 8;
    atomicAdd(dst + 0, a.x * rn); atomicAdd(dst + 1, a.y * rn);
    atomicAdd(dst + 2, a.z * rn); atomicAdd(dst + 3, a.w * rn);
    atomicAdd(dst + 4, b.x * rn); atomicAdd(dst + 5, b.y * rn);
    atomicAdd(dst + 6, b.z * rn); atomicAdd(dst + 7, b.w * rn);
    if (lane == 0) atomicAdd(&scnt[lab], 1);
  }
  __syncthreads();
  for (int i = threadIdx.x; i < C_CLS * D_DIM; i += 256) atomicAdd(&gseg[i], sseg[i]);
  if (threadIdx.x < C_CLS) atomicAdd(&gcnt[threadIdx.x], scnt[threadIdx.x]);
}

// ---------------- kernel 2: EMA prototype update + per-class weights ----------------
__global__ __launch_bounds__(128) void k_proto(const float* __restrict__ protos,
                                               const unsigned char* __restrict__ pinit,
                                               const int* __restrict__ pcount,
                                               const float* __restrict__ pvar,
                                               const int* __restrict__ step_p,
                                               char* ws) {
  __shared__ float s_n[C_CLS], s_f[C_CLS], s_q[C_CLS], s_w[C_CLS];
  __shared__ float sh_nv, sh_mf, sh_mq, sh_mw;
  float* seg     = (float*)(ws + OFF_SEG);
  float* protosn = (float*)(ws + OFF_PROTOSN);
  _Float16* ph   = (_Float16*)(ws + OFF_PROTOH);
  int*   cnt     = (int*)  (ws + OFF_CNT);
  float* wv      = (float*)(ws + OFF_WV);
  float* vf      = (float*)(ws + OFF_VALID);
  float* scal    = (float*)(ws + OFF_SCAL);

  const float progress = fminf(1.0f, (float)step_p[0] / 2000.0f); // WARMUP*10
  const float mom = 0.99f + (0.999f - 0.99f) * progress;
  const int c = threadIdx.x;

  float freq_raw = 0.f, qual_raw = 0.f; int validc = 0;
  if (c < C_CLS) {
    const int   nc   = cnt[c];
    const bool  has  = nc > 0;
    const bool  init = pinit[c] != 0;   // jnp.bool_ -> 1 byte
    const float den  = fmaxf((float)nc, 1.0f);
    float ss_np = 0.f, ss_df = 0.f;
    for (int d = 0; d < D_DIM; ++d) {
      const float mean = seg[c * D_DIM + d] / den;
      const float pr   = protos[c * D_DIM + d];
      const float ema  = mom * pr + (1.0f - mom) * mean;
      const float np   = has ? (init ? ema : mean) : pr;
      const float df   = mean - pr;
      ss_df += df * df; ss_np += np * np;
      seg[c * D_DIM + d] = np;            // stash new_proto in seg
    }
    const float mag = sqrtf(ss_df);
    const bool  upd = has && init;
    const float nvar = upd ? (0.99f * pvar[c] + 0.01f * mag) : pvar[c];
    const int   ncnt = pcount[c] + (has ? 1 : 0);
    validc = (init || has) ? 1 : 0;
    const float inorm = 1.f / fmaxf(sqrtf(ss_np), 1e-12f);
    for (int d = 0; d < D_DIM; ++d) {
      const float pn = seg[c * D_DIM + d] * inorm;
      protosn[c * D_DIM + d] = pn;
      ph[c * D_DIM + d] = (_Float16)pn;
    }
    freq_raw = 1.f / fmaxf((float)ncnt, 1.0f);
    qual_raw = 1.f / fmaxf(nvar, 1e-4f);
    const float nfl = (float)nc * (float)validc;
    s_n[c] = nfl; s_f[c] = nfl * freq_raw; s_q[c] = nfl * qual_raw;
    vf[c] = (float)validc;
  } else if (c < C_PAD) {
    for (int d = 0; d < D_DIM; ++d) ph[c * D_DIM + d] = (_Float16)0.f; // zero pad cols
    wv[c] = 0.f; vf[c] = 0.f;
  }
  __syncthreads();
  if (c == 0) {
    float sn = 0.f, sf = 0.f, sq = 0.f;
    for (int i = 0; i < C_CLS; ++i) { sn += s_n[i]; sf += s_f[i]; sq += s_q[i]; }
    const float nv = fmaxf(sn, 1e-8f);
    sh_nv = nv; sh_mf = sf / nv; sh_mq = sq / nv;
    scal[2] = 1.0f / nv;
  }
  __syncthreads();
  float wcl = 0.f;
  if (c < C_CLS) {
    const float fr = fminf(freq_raw / fmaxf(sh_mf, 1e-8f), 5.0f);
    const float ql = fminf(qual_raw / fmaxf(sh_mq, 1e-8f), 5.0f);
    wcl = fr * ql;
    s_w[c] = s_n[c] * wcl;
  }
  __syncthreads();
  if (c == 0) {
    float sw = 0.f;
    for (int i = 0; i < C_CLS; ++i) sw += s_w[i];
    sh_mw = sw / sh_nv;
  }
  __syncthreads();
  if (c < C_CLS) wv[c] = validc ? (wcl / fmaxf(sh_mw, 1e-8f)) : 0.f;
}

// ---------------- kernel 3: prototype repulsion ----------------
__global__ __launch_bounds__(256) void k_rep(char* ws) {
  __shared__ float sh_s[256], sh_n[256];
  const float* pn = (const float*)(ws + OFF_PROTOSN);
  const float* vf = (const float*)(ws + OFF_VALID);
  float* scal = (float*)(ws + OFF_SCAL);
  float s = 0.f, np = 0.f;
  for (int idx = threadIdx.x; idx < C_CLS * C_CLS; idx += 256) {
    const int i = idx / C_CLS, j = idx % C_CLS;
    if (j > i && vf[i] > 0.f && vf[j] > 0.f) {
      float dot = 0.f;
      for (int d = 0; d < D_DIM; ++d) dot += pn[i * D_DIM + d] * pn[j * D_DIM + d];
      s += fmaxf(dot, 0.f); np += 1.f;
    }
  }
  sh_s[threadIdx.x] = s; sh_n[threadIdx.x] = np;
  __syncthreads();
  if (threadIdx.x == 0) {
    float ts = 0.f, tn = 0.f;
    for (int i = 0; i < 256; ++i) { ts += sh_s[i]; tn += sh_n[i]; }
    scal[1] = (tn > 0.f) ? (ts / fmaxf(tn, 1.f)) : 0.f;
  }
}

// ---------------- kernel 4: WMMA logits + fused weighted CE ----------------
// grid 512 x 256: 8 waves/block, 4 16-row tiles per wave, 48 wmma per tile.
__global__ __launch_bounds__(256) void k_pull(const float* __restrict__ feats,
                                              const int* __restrict__ labels,
                                              char* ws) {
  const _Float16* ph  = (const _Float16*)(ws + OFF_PROTOH);
  const float* rnorm  = (const float*)(ws + OFF_RNORM);
  const float* wv     = (const float*)(ws + OFF_WV);
  float* scal         = (float*)(ws + OFF_SCAL);

  const int lane = threadIdx.x & 31;
  const int wav  = threadIdx.x >> 5;
  const int h    = lane >> 4;     // half: selects K sub-range
  const int ln   = lane & 15;     // row (A) / col (B,D) within tile

  float partial = 0.f;
  for (int it = 0; it < 4; ++it) {
    const int tile = blockIdx.x * 32 + wav * 4 + it;
    const int m0 = tile * 16;
    v8f acc[6] = {};
    const float* Ab = feats + (size_t)(m0 + ln) * D_DIM;
#pragma unroll
    for (int s = 0; s < 8; ++s) {
      const int kb = 32 * s + 16 * h;
      const float4 p0 = *(const float4*)(Ab + kb);
      const float4 p1 = *(const float4*)(Ab + kb + 4);
      const float4 q0 = *(const float4*)(Ab + kb + 8);
      const float4 q1 = *(const float4*)(Ab + kb + 12);
      v16h af;
      af[0]=(_Float16)p0.x; af[1]=(_Float16)p0.y; af[2]=(_Float16)p0.z; af[3]=(_Float16)p0.w;
      af[4]=(_Float16)p1.x; af[5]=(_Float16)p1.y; af[6]=(_Float16)p1.z; af[7]=(_Float16)p1.w;
      af[8]=(_Float16)q0.x; af[9]=(_Float16)q0.y; af[10]=(_Float16)q0.z; af[11]=(_Float16)q0.w;
      af[12]=(_Float16)q1.x; af[13]=(_Float16)q1.y; af[14]=(_Float16)q1.z; af[15]=(_Float16)q1.w;
#pragma unroll
      for (int t = 0; t < 6; ++t) {
        const v16h bf = *(const v16h*)(ph + ((size_t)(16 * t + ln)) * D_DIM + 32 * s + 16 * h);
        acc[t] = __builtin_amdgcn_wmma_f32_16x16x32_f16(
            false, af, false, bf, (short)0, acc[t], false, false);
      }
    }
    // fused masked log-softmax + weighted CE. D layout: vgpr v, half h -> row v+8h; col 16t+ln.
#pragma unroll
    for (int v = 0; v < 8; ++v) {
      const int grow = m0 + v + 8 * h;
      const float scale = rnorm[grow] * TAU_INV;
      float lg[6];
      float mx = -3.0e38f;
#pragma unroll
      for (int t = 0; t < 6; ++t) {
        lg[t] = acc[t][v] * scale;
        if (16 * t + ln < C_CLS) mx = fmaxf(mx, lg[t]);
      }
      mx = hred_max(mx);
      float se = 0.f;
#pragma unroll
      for (int t = 0; t < 6; ++t)
        if (16 * t + ln < C_CLS) se += expf(lg[t] - mx);
      se = hred_sum(se);
      const float lse = mx + logf(se);
      const int lab = labels[grow];
      const float wl = wv[lab];
#pragma unroll
      for (int t = 0; t < 6; ++t)
        if (16 * t + ln == lab) partial -= wl * lg[t];
      if (ln == 0) partial += wl * lse;
    }
  }
  partial = wred_sum(partial);
  if (lane == 0) atomicAdd(&scal[0], partial);
}

// ---------------- kernel 5: finalize ----------------
__global__ void k_final(const char* ws, float* out) {
  const float* scal = (const float*)(ws + OFF_SCAL);
  out[0] = scal[0] * scal[2] + REPC * scal[1];
}

extern "C" void kernel_launch(void* const* d_in, const int* in_sizes, int n_in,
                              void* d_out, int out_size, void* d_ws, size_t ws_size,
                              hipStream_t stream) {
  const float* feats          = (const float*)d_in[0];
  const int*   labels         = (const int*)d_in[1];
  const float* protos         = (const float*)d_in[2];
  const unsigned char* pinit  = (const unsigned char*)d_in[3];
  const int*   pcount         = (const int*)d_in[4];
  const float* pvar           = (const float*)d_in[5];
  const int*   step_p         = (const int*)d_in[6];
  float* out = (float*)d_out;
  char*  ws  = (char*)d_ws;

  k_zero<<<1, 256, 0, stream>>>(ws);
  const size_t lds1 = (size_t)C_CLS * D_DIM * 4 + C_PAD * 4;
  k_accum<<<256, 256, lds1, stream>>>(feats, labels, ws);
  k_proto<<<1, 128, 0, stream>>>(protos, pinit, pcount, pvar, step_p, ws);
  k_rep<<<1, 256, 0, stream>>>(ws);
  k_pull<<<512, 256, 0, stream>>>(feats, labels, ws);
  k_final<<<1, 1, 0, stream>>>(ws, out);
}
